// StackResRNN_34376918237767
// MI455X (gfx1250) — matrix operand order repaired and checked by reading, not compiled, not run on hardware
//
#include <hip/hip_runtime.h>
#include <hip/hip_bf16.h>

// ---------------------------------------------------------------------------
// StackResRNN on MI455X (gfx1250): WMMA f16 (f32 accum) persistent recurrence
//   B=32, T=96, D=64, H=768
// Dual-accumulator WMMA loops (breaks the f16-WMMA RAW hazard chain) +
// global_prefetch on the weight stream.
// ---------------------------------------------------------------------------

#define BB 32
#define TT 96
#define DD 64
#define HH 768

typedef __attribute__((ext_vector_type(8)))  _Float16 v8h;
typedef __attribute__((ext_vector_type(16))) _Float16 v16h;
typedef __attribute__((ext_vector_type(8)))  float    v8f;

// ---- workspace layout (bytes, all offsets 256-aligned) --------------------
static const size_t OFF_XH   = 0;                       // f16 [T*B][D]
static const size_t OFF_XPRE = 393216;                  // f32 [T*B][H]
static const size_t OFF_L1W  = 9830400;                 // f16 [768][64]
static const size_t OFF_L1U  = 9928704;                 // f16 [768][768]
static const size_t OFF_L2W  = 11108352;
static const size_t OFF_L2U  = 12288000;
static const size_t OFF_L3W  = 13467648;
static const size_t OFF_L3U  = 14647296;
static const size_t OFF_L4W  = 15826944;
static const size_t OFF_L4U  = 17006592;
static const size_t OFF_L4V  = 18186240;
static const size_t OFF_G1   = 19365888;                // f16 [768][1536]
static const size_t OFF_G2   = 21725184;                // f16 [768][1536]
static const size_t OFF_G3   = 24084480;                // f16 [768][2304]
static const size_t OFF_O3   = 27623424;                // f16 [64][768]
static const size_t OFF_O4   = 27721728;                // f16 [64][768]
static const size_t OFF_H    = 27820032;                // f16 h[2 banks][4][32*768]
static const size_t OFF_Z    = 28213248;                // f16 z[3][32*768]
static const size_t OFF_SYNC = 28360704;                // 2 x u32 barrier state
static const size_t STATE_BYTES = (size_t)BB * HH * 2;  // 49152

// ---------------------------------------------------------------------------
// WMMA helpers. Operand k order used by this code, A and B alike (16-bit A 16x32, B 32x16):
//   A lane<16 : M=lane,    K = kb..kb+15 (contiguous)
//   A lane>=16: M=lane-16, K = kb+16..kb+31
//   B lane<16 : N=lane,    K = kb..kb+15 (contiguous)  -> transposed weights
//   B lane>=16: N=lane-16, K = kb+16..kb+31
// ---------------------------------------------------------------------------
__device__ __forceinline__ v16h cat8(v8h lo, v8h hi) {
    return __builtin_shufflevector(lo, hi, 0, 1, 2, 3, 4, 5, 6, 7,
                                           8, 9, 10, 11, 12, 13, 14, 15);
}

__device__ __forceinline__ v8f wmma_f16(v16h a, v16h b, v8f c) {
    return __builtin_amdgcn_wmma_f32_16x16x32_f16(false, a, false, b,
                                                  (short)0, c, false, false);
}

// C(16x16,f32) += A[m0:m0+16, 0:K] * Bt^T, split across two independent
// accumulators (c0 even K-blocks, c1 odd) so consecutive v_wmma are
// independent and the RAW-hazard NOPs vanish.  K must be a multiple of 64.
__device__ __forceinline__ void gemm_acc2(v8f& c0, v8f& c1,
                                          const _Float16* __restrict__ A, int lda,
                                          const _Float16* __restrict__ Bt, int ldb,
                                          int K, int m0, int n0, int lane) {
    const int l15 = lane & 15;
    const int hi  = lane >> 4;
    const _Float16* a = A  + (size_t)(m0 + l15) * lda + hi * 16;
    const _Float16* b = Bt + (size_t)(n0 + l15) * ldb + hi * 16;
#pragma unroll 2
    for (int k = 0; k < K; k += 64) {
        __builtin_prefetch(b + k + 512, 0, 1);   // weight stream ~1KB ahead
        v8h  a00 = *(const v8h*)(a + k);
        v8h  a01 = *(const v8h*)(a + k + 8);
        v16h bv0 = *(const v16h*)(b + k);
        v8h  a10 = *(const v8h*)(a + k + 32);
        v8h  a11 = *(const v8h*)(a + k + 40);
        v16h bv1 = *(const v16h*)(b + k + 32);
        c0 = wmma_f16(cat8(a00, a01), bv0, c0);
        c1 = wmma_f16(cat8(a10, a11), bv1, c1);
    }
}

__device__ __forceinline__ float sigmoidf_(float x) {
    return 1.0f / (1.0f + __expf(-x));
}

// ---------------------------------------------------------------------------
// Device-wide sync for the 16-WG persistent grid.
// Emits the CDNA5 cluster split-barrier (NOP when not clustered) and backs it
// with an atomic generation barrier for guaranteed correctness.
// ---------------------------------------------------------------------------
__device__ __forceinline__ void grid_sync(unsigned* cnt, unsigned* gen, int nblocks) {
    __threadfence();        // release our state-slice stores to device scope
    __syncthreads();
    if (threadIdx.x == 0) {
        asm volatile("s_barrier_signal -3" ::: "memory");   // cluster signal
        unsigned g = __hip_atomic_load(gen, __ATOMIC_ACQUIRE, __HIP_MEMORY_SCOPE_AGENT);
        unsigned a = __hip_atomic_fetch_add(cnt, 1u, __ATOMIC_ACQ_REL, __HIP_MEMORY_SCOPE_AGENT);
        if (a == (unsigned)(nblocks - 1)) {
            __hip_atomic_store(cnt, 0u, __ATOMIC_RELAXED, __HIP_MEMORY_SCOPE_AGENT);
            __hip_atomic_fetch_add(gen, 1u, __ATOMIC_RELEASE, __HIP_MEMORY_SCOPE_AGENT);
        } else {
            while (__hip_atomic_load(gen, __ATOMIC_ACQUIRE, __HIP_MEMORY_SCOPE_AGENT) == g) {
                __builtin_amdgcn_s_sleep(1);
            }
        }
    }
    asm volatile("s_barrier_wait -3" ::: "memory");         // cluster wait (all waves)
    __syncthreads();
    __threadfence();        // acquire other WGPs' state stores
}

// ---------------------------------------------------------------------------
// Prep kernels
// ---------------------------------------------------------------------------
__global__ __launch_bounds__(256) void k_convert_t(const float* __restrict__ src,
                                                   _Float16* __restrict__ dst,
                                                   int fin, int fout) {
    int idx = blockIdx.x * 256 + threadIdx.x;
    if (idx < fin * fout) {
        int k = idx / fout, n = idx % fout;
        dst[(size_t)n * fin + k] = (_Float16)src[idx];   // dst = W^T, f16
    }
}

// x [B,T,D] f32 -> xh [(t*32+b)][D] f16
__global__ __launch_bounds__(256) void k_convert_x(const float* __restrict__ x,
                                                   _Float16* __restrict__ xh) {
    int idx = blockIdx.x * 256 + threadIdx.x;
    if (idx < TT * BB * DD) {
        int r = idx / DD, d = idx % DD;
        int b = r & 31, t = r >> 5;
        xh[idx] = (_Float16)x[((size_t)b * TT + t) * DD + d];
    }
}

__global__ __launch_bounds__(256) void k_init_state(char* ws) {
    int idx = blockIdx.x * 256 + threadIdx.x;
    _Float16* h0 = (_Float16*)(ws + OFF_H);   // bank 0: 4 states
    _Float16* z  = (_Float16*)(ws + OFF_Z);   // 3 gates
    if (idx < 4 * BB * HH) h0[idx] = (_Float16)0.0f;
    if (idx < 3 * BB * HH) z[idx]  = (_Float16)1.0f;
    if (idx == 0) {
        unsigned* s = (unsigned*)(ws + OFF_SYNC);
        s[0] = 0u; s[1] = 0u;
    }
}

// Xpre[(t*32+b)][H] = x @ l1W + b_l1W + b_l1U   (f32)
__global__ __launch_bounds__(256) void k_xpre(const _Float16* __restrict__ xh,
                                              const _Float16* __restrict__ w1t,
                                              const float* __restrict__ b1w,
                                              const float* __restrict__ b1u,
                                              float* __restrict__ xpre) {
    const int lane = threadIdx.x & 31, wave = threadIdx.x >> 5;
    const int l15 = lane & 15, hi = lane >> 4;
    const int tile = blockIdx.x * 8 + wave;            // 9216 tiles total
    const int m0 = (tile / 48) * 16;                   // rows of [T*B]
    const int n0 = (tile % 48) * 16;                   // cols of H
    v8f c0 = {}, c1 = {};
    gemm_acc2(c0, c1, xh, DD, w1t, DD, DD, m0, n0, lane);
    const int n = n0 + l15;
    const float bn = b1w[n] + b1u[n];
#pragma unroll
    for (int r = 0; r < 8; ++r)
        xpre[(size_t)(m0 + hi * 8 + r) * HH + n] = c0[r] + c1[r] + bn;
}

// ---------------------------------------------------------------------------
// Persistent recurrence kernel: 16 WGs x 192 threads (6 waves)
// wave -> tile (m0 = 16*(wave&1), n0 = 48*blockIdx.x + 16*(wave>>1))
// ---------------------------------------------------------------------------
struct RecurParams {
    char*  ws;
    float* out;
    const float *b_l2w, *b_l2u, *b_l3w, *b_l3u;
    const float *b_l4w, *b_l4u, *b_l4v;
    const float *b_g1, *b_g2, *b_g3;
    const float *b_o3, *b_o4;
};

__global__ __launch_bounds__(192) void k_recur(RecurParams p) {
    const int lane = threadIdx.x & 31, wave = threadIdx.x >> 5;
    const int l15 = lane & 15, hi = lane >> 4;
    const int m0 = (wave & 1) * 16;
    const int n0 = blockIdx.x * 48 + (wave >> 1) * 16;
    const int n  = n0 + l15;

    char* ws = p.ws;
    const _Float16* Wl1u = (const _Float16*)(ws + OFF_L1U);
    const _Float16* Wl2w = (const _Float16*)(ws + OFF_L2W);
    const _Float16* Wl2u = (const _Float16*)(ws + OFF_L2U);
    const _Float16* Wl3w = (const _Float16*)(ws + OFF_L3W);
    const _Float16* Wl3u = (const _Float16*)(ws + OFF_L3U);
    const _Float16* Wl4w = (const _Float16*)(ws + OFF_L4W);
    const _Float16* Wl4u = (const _Float16*)(ws + OFF_L4U);
    const _Float16* Wl4v = (const _Float16*)(ws + OFF_L4V);
    const _Float16* G1t  = (const _Float16*)(ws + OFF_G1);
    const _Float16* G2t  = (const _Float16*)(ws + OFF_G2);
    const _Float16* G3t  = (const _Float16*)(ws + OFF_G3);
    const _Float16* O3t  = (const _Float16*)(ws + OFF_O3);
    const _Float16* O4t  = (const _Float16*)(ws + OFF_O4);
    const float*    Xpre = (const float*)(ws + OFF_XPRE);

    _Float16* hb[2][4];
#pragma unroll
    for (int bk = 0; bk < 2; ++bk)
#pragma unroll
        for (int i = 0; i < 4; ++i)
            hb[bk][i] = (_Float16*)(ws + OFF_H + (size_t)(bk * 4 + i) * STATE_BYTES);
    _Float16* z1 = (_Float16*)(ws + OFF_Z);
    _Float16* z2 = (_Float16*)(ws + OFF_Z + STATE_BYTES);
    _Float16* z3 = (_Float16*)(ws + OFF_Z + 2 * STATE_BYTES);
    unsigned* cnt = (unsigned*)(ws + OFF_SYNC);
    unsigned* gen = cnt + 1;

    const float b2 = p.b_l2w[n] + p.b_l2u[n];
    const float b3 = p.b_l3w[n] + p.b_l3u[n];
    const float b4 = p.b_l4w[n] + p.b_l4u[n] + p.b_l4v[n];
    const float bg1 = p.b_g1[n], bg2 = p.b_g2[n], bg3 = p.b_g3[n];

    for (int t = 0; t < TT; ++t) {
        const int pr = t & 1;
        const _Float16 *h1o = hb[pr][0], *h2o = hb[pr][1];
        const _Float16 *h3o = hb[pr][2], *h4o = hb[pr][3];
        _Float16 *h1n = hb[pr ^ 1][0], *h2n = hb[pr ^ 1][1];
        _Float16 *h3n = hb[pr ^ 1][2], *h4n = hb[pr ^ 1][3];

        // ---- P1: h1 = tanh(Xpre[t] + h1_old @ l1U) -------------------------
        {
            v8f c0, c1 = {};
#pragma unroll
            for (int r = 0; r < 8; ++r)
                c0[r] = Xpre[(size_t)(t * BB + m0 + hi * 8 + r) * HH + n];
            gemm_acc2(c0, c1, h1o, HH, Wl1u, HH, HH, m0, n0, lane);
#pragma unroll
            for (int r = 0; r < 8; ++r) {
                int idx = (m0 + hi * 8 + r) * HH + n;
                h1n[idx] = (_Float16)tanhf(c0[r] + c1[r]);
            }
        }
        grid_sync(cnt, gen, 16);

        // ---- P2: h2 = z1*tanh(h1@l2W + h2o@l2U + b) + (1-z1)*h2o ----------
        {
            v8f c0, c1 = {};
#pragma unroll
            for (int r = 0; r < 8; ++r) c0[r] = b2;
            gemm_acc2(c0, c1, h1n, HH, Wl2w, HH, HH, m0, n0, lane);
            gemm_acc2(c0, c1, h2o, HH, Wl2u, HH, HH, m0, n0, lane);
#pragma unroll
            for (int r = 0; r < 8; ++r) {
                int idx = (m0 + hi * 8 + r) * HH + n;
                float z = (float)z1[idx], ho = (float)h2o[idx];
                h2n[idx] = (_Float16)(z * tanhf(c0[r] + c1[r]) + (1.0f - z) * ho);
            }
        }
        grid_sync(cnt, gen, 16);

        // ---- P3: h3 = z2*tanh(h2@l3W + h3o@l3U + b) + (1-z2)*h3o ----------
        {
            v8f c0, c1 = {};
#pragma unroll
            for (int r = 0; r < 8; ++r) c0[r] = b3;
            gemm_acc2(c0, c1, h2n, HH, Wl3w, HH, HH, m0, n0, lane);
            gemm_acc2(c0, c1, h3o, HH, Wl3u, HH, HH, m0, n0, lane);
#pragma unroll
            for (int r = 0; r < 8; ++r) {
                int idx = (m0 + hi * 8 + r) * HH + n;
                float z = (float)z2[idx], ho = (float)h3o[idx];
                h3n[idx] = (_Float16)(z * tanhf(c0[r] + c1[r]) + (1.0f - z) * ho);
            }
        }
        grid_sync(cnt, gen, 16);

        // ---- P4: h4 = z3*tanh(h3@l4W + h4o@l4U + h1@l4V + b) + (1-z3)*h4o -
        {
            v8f c0, c1 = {};
#pragma unroll
            for (int r = 0; r < 8; ++r) c0[r] = b4;
            gemm_acc2(c0, c1, h3n, HH, Wl4w, HH, HH, m0, n0, lane);
            gemm_acc2(c0, c1, h4o, HH, Wl4u, HH, HH, m0, n0, lane);
            gemm_acc2(c0, c1, h1n, HH, Wl4v, HH, HH, m0, n0, lane);
#pragma unroll
            for (int r = 0; r < 8; ++r) {
                int idx = (m0 + hi * 8 + r) * HH + n;
                float z = (float)z3[idx], ho = (float)h4o[idx];
                h4n[idx] = (_Float16)(z * tanhf(c0[r] + c1[r]) + (1.0f - z) * ho);
            }
        }
        grid_sync(cnt, gen, 16);

        // ---- P5: gates (z1,z2,z3) + output slice --------------------------
        {
            // g3 = [h3,h4,h1] @ gate3  (computed once; reused for z3 and z2)
            v8f c0, c1 = {};
#pragma unroll
            for (int r = 0; r < 8; ++r) c0[r] = bg3;
            gemm_acc2(c0, c1, h3n, HH, G3t,        2304, HH, m0, n0, lane);
            gemm_acc2(c0, c1, h4n, HH, G3t + 768,  2304, HH, m0, n0, lane);
            gemm_acc2(c0, c1, h1n, HH, G3t + 1536, 2304, HH, m0, n0, lane);
            float s3[8];
#pragma unroll
            for (int r = 0; r < 8; ++r) s3[r] = sigmoidf_(c0[r] + c1[r]);

            // z1 = sigmoid([h1,h2] @ gate1)
#pragma unroll
            for (int r = 0; r < 8; ++r) { c0[r] = bg1; c1[r] = 0.0f; }
            gemm_acc2(c0, c1, h1n, HH, G1t,        1536, HH, m0, n0, lane);
            gemm_acc2(c0, c1, h2n, HH, G1t + 768,  1536, HH, m0, n0, lane);
#pragma unroll
            for (int r = 0; r < 8; ++r) {
                int idx = (m0 + hi * 8 + r) * HH + n;
                z1[idx] = (_Float16)sigmoidf_(c0[r] + c1[r]);
            }

            // z2 = sigmoid([h2,h3] @ gate2) * s3 ;  z3 = s3
#pragma unroll
            for (int r = 0; r < 8; ++r) { c0[r] = bg2; c1[r] = 0.0f; }
            gemm_acc2(c0, c1, h2n, HH, G2t,        1536, HH, m0, n0, lane);
            gemm_acc2(c0, c1, h3n, HH, G2t + 768,  1536, HH, m0, n0, lane);
#pragma unroll
            for (int r = 0; r < 8; ++r) {
                int idx = (m0 + hi * 8 + r) * HH + n;
                z2[idx] = (_Float16)(sigmoidf_(c0[r] + c1[r]) * s3[r]);
                z3[idx] = (_Float16)s3[r];
            }

            // out[b,t,:] = h3 @ l3_h2o + h4 @ l4_h2o + biases   (D=64 -> 8 tiles)
            if (blockIdx.x < 8 && wave == 0) {
                const int m0o = (blockIdx.x >> 2) * 16;
                const int n0o = (blockIdx.x & 3) * 16;
                const int no  = n0o + l15;
                v8f d0, d1 = {};
                const float bo = p.b_o3[no] + p.b_o4[no];
#pragma unroll
                for (int r = 0; r < 8; ++r) d0[r] = bo;
                gemm_acc2(d0, d1, h3n, HH, O3t, HH, HH, m0o, n0o, lane);
                gemm_acc2(d0, d1, h4n, HH, O4t, HH, HH, m0o, n0o, lane);
#pragma unroll
                for (int r = 0; r < 8; ++r) {
                    int b = m0o + hi * 8 + r;          // batch row
                    p.out[((size_t)b * TT + t) * DD + no] = d0[r] + d1[r];
                }
            }
        }
        grid_sync(cnt, gen, 16);
    }
}

// ---------------------------------------------------------------------------
// Host launch
// ---------------------------------------------------------------------------
extern "C" void kernel_launch(void* const* d_in, const int* in_sizes, int n_in,
                              void* d_out, int out_size, void* d_ws, size_t ws_size,
                              hipStream_t stream) {
    (void)in_sizes; (void)n_in; (void)out_size; (void)ws_size;
    char* ws = (char*)d_ws;

    struct Job { int in; size_t off; int fin; int fout; };
    const Job jobs[] = {
        {1,  OFF_L1W, 64,   768}, {3,  OFF_L1U, 768, 768},
        {5,  OFF_L2W, 768,  768}, {7,  OFF_L2U, 768, 768},
        {9,  OFF_L3W, 768,  768}, {11, OFF_L3U, 768, 768},
        {13, OFF_L4W, 768,  768}, {15, OFF_L4U, 768, 768},
        {17, OFF_L4V, 768,  768},
        {19, OFF_G1,  1536, 768}, {21, OFF_G2, 1536, 768},
        {23, OFF_G3,  2304, 768},
        {25, OFF_O3,  768,  64},  {27, OFF_O4, 768, 64},
    };
    for (const Job& j : jobs) {
        int nel = j.fin * j.fout;
        k_convert_t<<<(nel + 255) / 256, 256, 0, stream>>>(
            (const float*)d_in[j.in], (_Float16*)(ws + j.off), j.fin, j.fout);
    }

    k_convert_x<<<(TT * BB * DD + 255) / 256, 256, 0, stream>>>(
        (const float*)d_in[0], (_Float16*)(ws + OFF_XH));

    k_init_state<<<(4 * BB * HH + 255) / 256, 256, 0, stream>>>(ws);

    k_xpre<<<(TT * BB / 16) * 48 / 8, 256, 0, stream>>>(
        (const _Float16*)(ws + OFF_XH), (const _Float16*)(ws + OFF_L1W),
        (const float*)d_in[2], (const float*)d_in[4], (float*)(ws + OFF_XPRE));

    RecurParams rp;
    rp.ws = ws;
    rp.out = (float*)d_out;
    rp.b_l2w = (const float*)d_in[6];  rp.b_l2u = (const float*)d_in[8];
    rp.b_l3w = (const float*)d_in[10]; rp.b_l3u = (const float*)d_in[12];
    rp.b_l4w = (const float*)d_in[14]; rp.b_l4u = (const float*)d_in[16];
    rp.b_l4v = (const float*)d_in[18];
    rp.b_g1  = (const float*)d_in[20]; rp.b_g2 = (const float*)d_in[22];
    rp.b_g3  = (const float*)d_in[24];
    rp.b_o3  = (const float*)d_in[26]; rp.b_o4 = (const float*)d_in[28];

    k_recur<<<16, 192, 0, stream>>>(rp);
}
